// MoELayer_24352464569735
// MI455X (gfx1250) — compile-verified, matrix operand order checked by reading, not run on hardware
//
#include <hip/hip_runtime.h>
#include <hip/hip_bf16.h>

typedef __attribute__((ext_vector_type(16))) __bf16 v16bf;
typedef __attribute__((ext_vector_type(8)))  __bf16 v8bf;
typedef __attribute__((ext_vector_type(4)))  __bf16 v4bf;
typedef __attribute__((ext_vector_type(8)))  float  v8f;

#define NT   4096   // tokens
#define DM   1024   // d_model
#define NE   16     // experts
#define FF   512    // expert ffn
#define FFS  1024   // shared ffn
#define CAP  1024   // capacity

// ---------- helpers ----------

static __device__ inline v16bf make_frag(const __bf16* p0, const __bf16* p1) {
    v8bf lo = *(const v8bf*)p0;
    v8bf hi = *(const v8bf*)p1;
    v16bf r;
#pragma unroll
    for (int i = 0; i < 8; ++i) { r[i] = lo[i]; r[i + 8] = hi[i]; }
    return r;
}

static __device__ inline v8f vzero8() {
    v8f z = {0.f, 0.f, 0.f, 0.f, 0.f, 0.f, 0.f, 0.f};
    return z;
}

static __device__ inline float fast_rcp(float x) { return __builtin_amdgcn_rcpf(x); }

// ---------- small kernels ----------

__global__ void k_zero_cnt(int* cnt) {
    if (threadIdx.x < NE) cnt[threadIdx.x] = 0;
}

__global__ void k_f32_to_bf16(const float* __restrict__ src, __bf16* __restrict__ dst, int n4) {
    int i = blockIdx.x * blockDim.x + threadIdx.x;
    if (i >= n4) return;
    float4 f = ((const float4*)src)[i];
    v4bf o;
    o[0] = (__bf16)f.x; o[1] = (__bf16)f.y; o[2] = (__bf16)f.z; o[3] = (__bf16)f.w;
    ((v4bf*)dst)[i] = o;
}

// transpose fp32 W[K][N] -> bf16 Wt[N][K]; one-time per call, amortized over all M-tiles
__global__ __launch_bounds__(256) void k_transpose_bf16(
    const float* __restrict__ W, __bf16* __restrict__ Wt,
    int K, int N, long wStride, long wtStride)
{
    const int b = blockIdx.z;
    W  += (size_t)b * wStride;
    Wt += (size_t)b * wtStride;
    __shared__ float tile[32][33];
    const int k0 = blockIdx.y * 32, n0 = blockIdx.x * 32;
    const int tx = threadIdx.x & 31;
    const int ry = (threadIdx.x >> 5) * 4;
#pragma unroll
    for (int r = ry; r < ry + 4; ++r)
        tile[r][tx] = W[(size_t)(k0 + r) * N + n0 + tx];
    __syncthreads();
#pragma unroll
    for (int r = ry; r < ry + 4; ++r)
        Wt[(size_t)(n0 + r) * K + k0 + tx] = (__bf16)tile[tx][r];
}

// one wave per token: lane&15 = expert, lane>>4 = K-half
__global__ void k_router(const float* __restrict__ hidden, const float* __restrict__ Wg,
                         float* __restrict__ wts, int* __restrict__ slots, int* __restrict__ cnt) {
    const int lane = threadIdx.x & 31;
    const int wid  = threadIdx.x >> 5;
    const int t    = blockIdx.x * 8 + wid;
    const int e    = lane & 15;
    const int half = lane >> 4;

    const float* hrow = hidden + (size_t)t * DM;
    float acc = 0.f;
    const int d0 = half * (DM / 2);
    for (int d = d0; d < d0 + DM / 2; ++d)
        acc += hrow[d] * Wg[d * NE + e];
    acc += __shfl_xor(acc, 16);   // both halves hold full logit for expert e

    float m = acc;
#pragma unroll
    for (int o = 8; o >= 1; o >>= 1) m = fmaxf(m, __shfl_xor(m, o));
    float p = __expf(acc - m);
    float s = p;
#pragma unroll
    for (int o = 8; o >= 1; o >>= 1) s += __shfl_xor(s, o);
    float prob = p * fast_rcp(s);

    float v1 = prob; int i1 = e;
#pragma unroll
    for (int o = 8; o >= 1; o >>= 1) {
        float ov = __shfl_xor(v1, o); int oi = __shfl_xor(i1, o);
        if (ov > v1 || (ov == v1 && oi < i1)) { v1 = ov; i1 = oi; }
    }
    float pv = (e == i1) ? -1.0f : prob;
    float v2 = pv; int i2 = e;
#pragma unroll
    for (int o = 8; o >= 1; o >>= 1) {
        float ov = __shfl_xor(v2, o); int oi = __shfl_xor(i2, o);
        if (ov > v2 || (ov == v2 && oi < i2)) { v2 = ov; i2 = oi; }
    }

    if (lane == 0) {
        float inv = fast_rcp(v1 + v2);
        int p0 = atomicAdd(&cnt[i1], 1);
        int p1 = atomicAdd(&cnt[i2], 1);
        slots[2 * t]     = (p0 < CAP) ? i1 * CAP + p0 : -1;
        slots[2 * t + 1] = (p1 < CAP) ? i2 * CAP + p1 : -1;
        wts[2 * t]     = v1 * inv;
        wts[2 * t + 1] = v2 * inv;
    }
}

// one block per (token,k) pair: copy bf16 row into expert buffer
__global__ void k_dispatch(const __bf16* __restrict__ hbf, const int* __restrict__ slots,
                           __bf16* __restrict__ xb) {
    const int pair = blockIdx.x;
    const int s = slots[pair];
    if (s < 0) return;
    const int t = pair >> 1;
    const int c = threadIdx.x * 4;
    *(v4bf*)(xb + (size_t)s * DM + c) = *(const v4bf*)(hbf + (size_t)t * DM + c);
}

// ---------- WMMA GEMM kernels ----------
// Block tile 64(M) x 128(N), BK=32, 256 threads = 8 waves (2x4), wave = 32x32 = 2x2 wmma tiles.
// A: bf16 row-major [M][K]. W: bf16 pre-transposed [N][K]. Software-pipelined global->reg->LDS.

__global__ __launch_bounds__(256) void k_swiglu_gemm(
    const __bf16* __restrict__ A, const __bf16* __restrict__ W1t, const __bf16* __restrict__ W3t,
    __bf16* __restrict__ H, int M, int K, int N,
    long aStride, long wStride, long hStride, const int* __restrict__ counts)
{
    const int b     = blockIdx.z;
    const int mbase = blockIdx.y * 64;
    const int nbase = blockIdx.x * 128;
    int mvalid = M;
    if (counts) { int c = counts[b]; mvalid = c < M ? c : M; }
    if (mbase >= mvalid) return;

    A   += (size_t)b * aStride;
    W1t += (size_t)b * wStride;
    W3t += (size_t)b * wStride;
    H   += (size_t)b * hStride;

    __shared__ __align__(16) __bf16 sA [64 * 32];
    __shared__ __align__(16) __bf16 sB1[128 * 32];   // [n][k]
    __shared__ __align__(16) __bf16 sB3[128 * 32];

    const int tid  = threadIdx.x;
    const int lane = tid & 31, wid = tid >> 5;
    const int waveM = wid >> 2, waveN = wid & 3;
    const int lrow = lane & 15, half = lane >> 4;
    const int mw = waveM * 32, nw = waveN * 32;

    v8f acc1[2][2], acc3[2][2];
#pragma unroll
    for (int mi = 0; mi < 2; ++mi)
#pragma unroll
        for (int nj = 0; nj < 2; ++nj) { acc1[mi][nj] = vzero8(); acc3[mi][nj] = vzero8(); }

    // A staging: 64x32 bf16 = 1 uint4/thread. B staging: 128x32 bf16 = 2 uint4/thread each.
    const int arow = tid >> 2;          // 0..63
    const int acol = (tid & 3) * 8;     // 0,8,16,24
    const int brow0 = tid >> 1;                 // chunk (tid + 0)   -> row 0..127
    const int bcol0 = (tid & 1) * 8;            //                     col 0 or 8
    const int brow1 = (tid + 256) >> 1;         // chunk (tid + 256) -> row 128.. wait: see below
    // chunk id c in [0,512): row = c>>2, col = (c&3)*8
    const int c0r = tid >> 2, c0c = (tid & 3) * 8;
    const int c1r = (tid + 256) >> 2, c1c = ((tid + 256) & 3) * 8;
    (void)brow0; (void)bcol0; (void)brow1;

    uint4 pa, pb1[2], pb3[2];
    {
        pa = *(const uint4*)(A + (size_t)(mbase + arow) * K + acol);
        pb1[0] = *(const uint4*)(W1t + (size_t)(nbase + c0r) * K + c0c);
        pb1[1] = *(const uint4*)(W1t + (size_t)(nbase + c1r) * K + c1c);
        pb3[0] = *(const uint4*)(W3t + (size_t)(nbase + c0r) * K + c0c);
        pb3[1] = *(const uint4*)(W3t + (size_t)(nbase + c1r) * K + c1c);
    }

    for (int k0 = 0; k0 < K; k0 += 32) {
        *(uint4*)(sA  + arow * 32 + acol) = pa;
        *(uint4*)(sB1 + c0r * 32 + c0c)   = pb1[0];
        *(uint4*)(sB1 + c1r * 32 + c1c)   = pb1[1];
        *(uint4*)(sB3 + c0r * 32 + c0c)   = pb3[0];
        *(uint4*)(sB3 + c1r * 32 + c1c)   = pb3[1];
        __syncthreads();

        const int kn = k0 + 32;
        if (kn < K) {   // prefetch next tile while WMMAs run
            pa = *(const uint4*)(A + (size_t)(mbase + arow) * K + kn + acol);
            pb1[0] = *(const uint4*)(W1t + (size_t)(nbase + c0r) * K + kn + c0c);
            pb1[1] = *(const uint4*)(W1t + (size_t)(nbase + c1r) * K + kn + c1c);
            pb3[0] = *(const uint4*)(W3t + (size_t)(nbase + c0r) * K + kn + c0c);
            pb3[1] = *(const uint4*)(W3t + (size_t)(nbase + c1r) * K + kn + c1c);
        }

        v16bf af[2], b1f[2], b3f[2];
#pragma unroll
        for (int mi = 0; mi < 2; ++mi) {
            const __bf16* p = sA + (mw + mi * 16 + lrow) * 32;
            af[mi] = make_frag(p + half * 8, p + 16 + half * 8);
        }
#pragma unroll
        for (int nj = 0; nj < 2; ++nj) {
            const __bf16* p1 = sB1 + (nw + nj * 16 + lrow) * 32;
            b1f[nj] = make_frag(p1 + half * 8, p1 + 16 + half * 8);
            const __bf16* p3 = sB3 + (nw + nj * 16 + lrow) * 32;
            b3f[nj] = make_frag(p3 + half * 8, p3 + 16 + half * 8);
        }
#pragma unroll
        for (int mi = 0; mi < 2; ++mi)
#pragma unroll
            for (int nj = 0; nj < 2; ++nj) {
                acc1[mi][nj] = __builtin_amdgcn_wmma_f32_16x16x32_bf16(
                    false, af[mi], false, b1f[nj], (short)0, acc1[mi][nj], false, false);
                acc3[mi][nj] = __builtin_amdgcn_wmma_f32_16x16x32_bf16(
                    false, af[mi], false, b3f[nj], (short)0, acc3[mi][nj], false, false);
            }
        __syncthreads();
    }

    // epilogue: h = silu(g) * u -> bf16   (fast rcp: g * rcp(1+exp(-g)) * u)
#pragma unroll
    for (int mi = 0; mi < 2; ++mi)
#pragma unroll
        for (int nj = 0; nj < 2; ++nj)
#pragma unroll
            for (int i = 0; i < 8; ++i) {
                float g = acc1[mi][nj][i];
                float u = acc3[mi][nj][i];
                float hv = g * fast_rcp(1.f + __expf(-g)) * u;
                int grow = mbase + mw + mi * 16 + half * 8 + i;
                int gcol = nbase + nw + nj * 16 + lrow;
                H[(size_t)grow * N + gcol] = (__bf16)hv;
            }
}

__global__ __launch_bounds__(256) void k_gemm(
    const __bf16* __restrict__ A, const __bf16* __restrict__ Wt, float* __restrict__ O,
    int M, int K, int N, long aStride, long wStride, long oStride,
    const int* __restrict__ counts)
{
    const int b     = blockIdx.z;
    const int mbase = blockIdx.y * 64;
    const int nbase = blockIdx.x * 128;
    int mvalid = M;
    if (counts) { int c = counts[b]; mvalid = c < M ? c : M; }
    if (mbase >= mvalid) return;

    A  += (size_t)b * aStride;
    Wt += (size_t)b * wStride;
    O  += (size_t)b * oStride;

    __shared__ __align__(16) __bf16 sA[64 * 32];
    __shared__ __align__(16) __bf16 sB[128 * 32];    // [n][k]

    const int tid  = threadIdx.x;
    const int lane = tid & 31, wid = tid >> 5;
    const int waveM = wid >> 2, waveN = wid & 3;
    const int lrow = lane & 15, half = lane >> 4;
    const int mw = waveM * 32, nw = waveN * 32;

    v8f acc[2][2];
#pragma unroll
    for (int mi = 0; mi < 2; ++mi)
#pragma unroll
        for (int nj = 0; nj < 2; ++nj) acc[mi][nj] = vzero8();

    const int arow = tid >> 2;
    const int acol = (tid & 3) * 8;
    const int c0r = tid >> 2, c0c = (tid & 3) * 8;
    const int c1r = (tid + 256) >> 2, c1c = ((tid + 256) & 3) * 8;

    uint4 pa, pb[2];
    pa = *(const uint4*)(A + (size_t)(mbase + arow) * K + acol);
    pb[0] = *(const uint4*)(Wt + (size_t)(nbase + c0r) * K + c0c);
    pb[1] = *(const uint4*)(Wt + (size_t)(nbase + c1r) * K + c1c);

    for (int k0 = 0; k0 < K; k0 += 32) {
        *(uint4*)(sA + arow * 32 + acol) = pa;
        *(uint4*)(sB + c0r * 32 + c0c)   = pb[0];
        *(uint4*)(sB + c1r * 32 + c1c)   = pb[1];
        __syncthreads();

        const int kn = k0 + 32;
        if (kn < K) {
            pa = *(const uint4*)(A + (size_t)(mbase + arow) * K + kn + acol);
            pb[0] = *(const uint4*)(Wt + (size_t)(nbase + c0r) * K + kn + c0c);
            pb[1] = *(const uint4*)(Wt + (size_t)(nbase + c1r) * K + kn + c1c);
        }

        v16bf af[2], bf[2];
#pragma unroll
        for (int mi = 0; mi < 2; ++mi) {
            const __bf16* p = sA + (mw + mi * 16 + lrow) * 32;
            af[mi] = make_frag(p + half * 8, p + 16 + half * 8);
        }
#pragma unroll
        for (int nj = 0; nj < 2; ++nj) {
            const __bf16* p = sB + (nw + nj * 16 + lrow) * 32;
            bf[nj] = make_frag(p + half * 8, p + 16 + half * 8);
        }
#pragma unroll
        for (int mi = 0; mi < 2; ++mi)
#pragma unroll
            for (int nj = 0; nj < 2; ++nj)
                acc[mi][nj] = __builtin_amdgcn_wmma_f32_16x16x32_bf16(
                    false, af[mi], false, bf[nj], (short)0, acc[mi][nj], false, false);
        __syncthreads();
    }

#pragma unroll
    for (int mi = 0; mi < 2; ++mi)
#pragma unroll
        for (int nj = 0; nj < 2; ++nj)
#pragma unroll
            for (int i = 0; i < 8; ++i) {
                int grow = mbase + mw + mi * 16 + half * 8 + i;
                int gcol = nbase + nw + nj * 16 + lrow;
                O[(size_t)grow * N + gcol] = acc[mi][nj][i];
            }
}

// out[t] (already = shared-expert result) += w0*Ob[slot0] + w1*Ob[slot1]
__global__ void k_combine(float* __restrict__ out, const float* __restrict__ Ob,
                          const int* __restrict__ slots, const float* __restrict__ wts) {
    const int t = blockIdx.x;
    const int c = threadIdx.x * 4;
    const int s0 = slots[2 * t], s1 = slots[2 * t + 1];
    const float w0 = wts[2 * t], w1 = wts[2 * t + 1];
    float4 acc = *(const float4*)(out + (size_t)t * DM + c);
    if (s0 >= 0) {
        float4 v = *(const float4*)(Ob + (size_t)s0 * DM + c);
        acc.x += w0 * v.x; acc.y += w0 * v.y; acc.z += w0 * v.z; acc.w += w0 * v.w;
    }
    if (s1 >= 0) {
        float4 v = *(const float4*)(Ob + (size_t)s1 * DM + c);
        acc.x += w1 * v.x; acc.y += w1 * v.y; acc.z += w1 * v.z; acc.w += w1 * v.w;
    }
    *(float4*)(out + (size_t)t * DM + c) = acc;
}

// ---------- launcher ----------

extern "C" void kernel_launch(void* const* d_in, const int* in_sizes, int n_in,
                              void* d_out, int out_size, void* d_ws, size_t ws_size,
                              hipStream_t stream) {
    const float* hidden = (const float*)d_in[0];
    const float* Wg  = (const float*)d_in[1];
    const float* w1  = (const float*)d_in[2];
    const float* w3  = (const float*)d_in[3];
    const float* w2  = (const float*)d_in[4];
    const float* ws1 = (const float*)d_in[5];
    const float* ws3 = (const float*)d_in[6];
    const float* ws2 = (const float*)d_in[7];
    float* out = (float*)d_out;

    char* ws = (char*)d_ws;
    size_t off = 0;
    auto alloc = [&](size_t bytes) -> void* {
        void* p = ws + off;
        off = (off + bytes + 255) & ~(size_t)255;
        return p;
    };

    int*    cnt   = (int*)   alloc(NE * sizeof(int));
    int*    slots = (int*)   alloc((size_t)NT * 2 * sizeof(int));
    float*  wts   = (float*) alloc((size_t)NT * 2 * sizeof(float));
    __bf16* hbf   = (__bf16*)alloc((size_t)NT * DM * 2);
    __bf16* xb    = (__bf16*)alloc((size_t)NE * CAP * DM * 2);
    __bf16* Hx    = (__bf16*)alloc((size_t)NE * CAP * FF * 2);
    float*  Ob    = (float*) alloc((size_t)NE * CAP * DM * 4);
    __bf16* Hs    = (__bf16*)alloc((size_t)NT * FFS * 2);
    __bf16* w1t   = (__bf16*)alloc((size_t)NE * DM * FF * 2);   // [e][FF][DM]
    __bf16* w3t   = (__bf16*)alloc((size_t)NE * DM * FF * 2);
    __bf16* w2t   = (__bf16*)alloc((size_t)NE * FF * DM * 2);   // [e][DM][FF]
    __bf16* ws1t  = (__bf16*)alloc((size_t)DM * FFS * 2);       // [FFS][DM]
    __bf16* ws3t  = (__bf16*)alloc((size_t)DM * FFS * 2);
    __bf16* ws2t  = (__bf16*)alloc((size_t)FFS * DM * 2);       // [DM][FFS]

    // --- one-time weight transpose+convert (amortized across all M tiles) ---
    { dim3 g(FF / 32, DM / 32, NE);
      k_transpose_bf16<<<g, 256, 0, stream>>>(w1, w1t, DM, FF, (long)DM * FF, (long)DM * FF);
      k_transpose_bf16<<<g, 256, 0, stream>>>(w3, w3t, DM, FF, (long)DM * FF, (long)DM * FF); }
    { dim3 g(DM / 32, FF / 32, NE);
      k_transpose_bf16<<<g, 256, 0, stream>>>(w2, w2t, FF, DM, (long)FF * DM, (long)FF * DM); }
    { dim3 g(FFS / 32, DM / 32, 1);
      k_transpose_bf16<<<g, 256, 0, stream>>>(ws1, ws1t, DM, FFS, 0, 0);
      k_transpose_bf16<<<g, 256, 0, stream>>>(ws3, ws3t, DM, FFS, 0, 0); }
    { dim3 g(DM / 32, FFS / 32, 1);
      k_transpose_bf16<<<g, 256, 0, stream>>>(ws2, ws2t, FFS, DM, 0, 0); }

    // --- router + dispatch ---
    k_zero_cnt<<<1, 32, 0, stream>>>(cnt);
    k_f32_to_bf16<<<(NT * DM / 4 + 255) / 256, 256, 0, stream>>>(hidden, hbf, NT * DM / 4);
    k_router<<<NT / 8, 256, 0, stream>>>(hidden, Wg, wts, slots, cnt);
    k_dispatch<<<NT * 2, 256, 0, stream>>>(hbf, slots, xb);

    // --- expert SwiGLU: [CAP x DM] @ [DM x FF] per expert ---
    dim3 g1(FF / 128, CAP / 64, NE);
    k_swiglu_gemm<<<g1, 256, 0, stream>>>(xb, w1t, w3t, Hx, CAP, DM, FF,
                                          (long)CAP * DM, (long)DM * FF, (long)CAP * FF, cnt);
    // --- expert down-proj: [CAP x FF] @ [FF x DM] ---
    dim3 g2(DM / 128, CAP / 64, NE);
    k_gemm<<<g2, 256, 0, stream>>>(Hx, w2t, Ob, CAP, FF, DM,
                                   (long)CAP * FF, (long)FF * DM, (long)CAP * DM, cnt);
    // --- shared SwiGLU: [NT x DM] @ [DM x FFS] ---
    dim3 g3(FFS / 128, NT / 64, 1);
    k_swiglu_gemm<<<g3, 256, 0, stream>>>(hbf, ws1t, ws3t, Hs, NT, DM, FFS, 0, 0, 0, nullptr);
    // --- shared down-proj straight into d_out: [NT x FFS] @ [FFS x DM] ---
    dim3 g4(DM / 128, NT / 64, 1);
    k_gemm<<<g4, 256, 0, stream>>>(Hs, ws2t, out, NT, FFS, DM, 0, 0, 0, nullptr);

    k_combine<<<NT, 256, 0, stream>>>(out, Ob, slots, wts);
}
